// CSDKM_66417374265458
// MI455X (gfx1250) — matrix operand order repaired and checked by reading, not compiled, not run on hardware
//
#include <hip/hip_runtime.h>
#include <hip/hip_bf16.h>

typedef __attribute__((ext_vector_type(16))) _Float16 v16h;
typedef __attribute__((ext_vector_type(8)))  float    v8f;
typedef __attribute__((ext_vector_type(4)))  float    v4f;
typedef __attribute__((ext_vector_type(4)))  int      v4i;
typedef _Float16 h16;

// Problem constants
#define KTOT 2816   // 256*9 (im2col c4) + 512 (c5_up)
#define MTOT 32768  // B*H*W = 8*64*64

// Workspace layout (bytes)
#define OFF_WCAT   0u          // 2816*256 f16 fragment-packed       (1,441,792 B)
#define OFF_WBIG   1572864u    // 256*512 f16 fragment-packed        (  262,144 B)
#define OFF_POOL4  2097152u    // 8*256*9 f32
#define OFF_POOL5  2359296u    // 8*512*9 f32
#define OFF_KERNS  2621440u    // 8*9*9 f32
#define OFF_BNPART 3145728u    // 2*256*256 f32 partial sums
#define OFF_BNSTAT 3670016u    // mu[256], inv_std[256]
#define OFF_FUSED  4194304u    // 32768*256 f32 NHWC
#define OFF_YBIG   37748736u   // 32768*512 f32 (y | fused_red)  ends at 100 MB

// Async global->LDS support (guarded: falls back to synchronous staging)
#if defined(__HIP_DEVICE_COMPILE__) && __has_builtin(__builtin_amdgcn_global_load_async_to_lds_b128)
#define GL_ASYNC 1
typedef __attribute__((address_space(1))) v4i* gasp;   // global int4*
typedef __attribute__((address_space(3))) v4i* lasp;   // LDS int4*
#if __has_builtin(__builtin_amdgcn_s_wait_asynccnt)
#define WAIT_ASYNC() __builtin_amdgcn_s_wait_asynccnt(0)
#else
#define WAIT_ASYNC() asm volatile("s_wait_asynccnt 0x0" ::: "memory")
#endif
#else
#define GL_ASYNC 0
#define WAIT_ASYNC()
#endif

// ---------------------------------------------------------------------------
// Weight packing into WMMA B-fragment layout:
// 32x16 f16 block -> 512 halfs; lane = (kk<16 ? nn : nn+16), pos = kk mod 16.
// ---------------------------------------------------------------------------
__global__ void k_pack_wcat(const float* __restrict__ wc4,   // (256,256,3,3)
                            const float* __restrict__ wc1,   // (256,512,1,1)
                            h16* __restrict__ wcat) {
  int g = blockIdx.x * 256 + threadIdx.x;      // 2816 blocks -> 720896 = KTOT*256
  int k = g % KTOT, n = g / KTOT;
  float v = (k < 2304) ? wc4[n * 2304 + k] : wc1[n * 512 + (k - 2304)];
  int kb = k >> 5, kk = k & 31, nb = n >> 4, nn = n & 15;
  int lane = (kk < 16) ? nn : nn + 16;
  int pos  = (kk < 16) ? kk : kk - 16;
  wcat[(size_t)(kb * 16 + nb) * 512 + lane * 16 + pos] = (h16)v;
}

__global__ void k_pack_wbig(const float* __restrict__ wtf,   // (256,256)
                            const float* __restrict__ wrs,   // (128,256)
                            const float* __restrict__ wpj,   // (256,128)
                            h16* __restrict__ wbig) {
  int g = blockIdx.x * 256 + threadIdx.x;      // 512 blocks -> 131072
  int k = g & 255, n = g >> 8;                 // k: fused channel, n: out col (0..511)
  float v;
  if (n < 256) {
    v = wtf[n * 256 + k];                      // y = fused @ w_to_fuse^T
  } else {                                     // Wcomb = w_proj @ w_reshape
    int o = n - 256; float s = 0.f;
    for (int f = 0; f < 128; ++f) s += wpj[o * 128 + f] * wrs[f * 256 + k];
    v = s;
  }
  int kb = k >> 5, kk = k & 31, nb = n >> 4, nn = n & 15;
  int lane = (kk < 16) ? nn : nn + 16;
  int pos  = (kk < 16) ? kk : kk - 16;
  wbig[(size_t)(kb * 32 + nb) * 512 + lane * 16 + pos] = (h16)v;
}

// ---------------------------------------------------------------------------
// Adaptive average pooling (3x3 regions, overlapping: rows [0,22),[21,43),[42,64))
// ---------------------------------------------------------------------------
__global__ void k_pool4(const float* __restrict__ c4, float* __restrict__ pool4) {
  int idx = blockIdx.x;                        // b*256*9 + c*9 + r
  int r = idx % 9, c = (idx / 9) & 255, b = idx / (9 * 256);
  int ri = r / 3, rj = r % 3;
  int h0 = (ri * 64) / 3, h1 = ((ri + 1) * 64 + 2) / 3;
  int w0 = (rj * 64) / 3, w1 = ((rj + 1) * 64 + 2) / 3;
  int nw = w1 - w0, n = (h1 - h0) * nw;
  const float* base = c4 + ((size_t)(b * 256 + c) << 12);
  float s = 0.f;
  for (int e = threadIdx.x; e < n; e += 64) {
    int hh = h0 + e / nw, ww = w0 + e % nw;
    s += base[(hh << 6) + ww];
  }
  __shared__ float red[64];
  red[threadIdx.x] = s; __syncthreads();
  for (int st = 32; st > 0; st >>= 1) {
    if (threadIdx.x < st) red[threadIdx.x] += red[threadIdx.x + st];
    __syncthreads();
  }
  if (threadIdx.x == 0) pool4[(size_t)(b * 256 + c) * 9 + r] = red[0] / (float)n;
}

__global__ void k_pool5(const float* __restrict__ c5, float* __restrict__ pool5) {
  int idx = blockIdx.x;                        // b*512*9 + c*9 + r
  int r = idx % 9, c = (idx / 9) & 511, b = idx / (9 * 512);
  int ri = r / 3, rj = r % 3;
  int h0 = (ri * 64) / 3, h1 = ((ri + 1) * 64 + 2) / 3;
  int w0 = (rj * 64) / 3, w1 = ((rj + 1) * 64 + 2) / 3;
  int nw = w1 - w0, n = (h1 - h0) * nw;
  const float* base = c5 + ((size_t)(b * 512 + c) << 10);
  float s = 0.f;
  for (int e = threadIdx.x; e < n; e += 64) {
    int hh = h0 + e / nw, ww = w0 + e % nw;
    s += base[((hh >> 1) << 5) + (ww >> 1)];   // nearest upsample of c5
  }
  __shared__ float red[64];
  red[threadIdx.x] = s; __syncthreads();
  for (int st = 32; st > 0; st >>= 1) {
    if (threadIdx.x < st) red[threadIdx.x] += red[threadIdx.x + st];
    __syncthreads();
  }
  if (threadIdx.x == 0) pool5[(size_t)(b * 512 + c) * 9 + r] = red[0] / (float)n;
}

// sim -> gate -> tiny MLP -> softmax kernel coefficients (per batch, region)
__global__ void k_sim_mlp(const float* __restrict__ pool4, const float* __restrict__ pool5,
                          const float* __restrict__ ws4, const float* __restrict__ ws5,
                          const float* __restrict__ w1, const float* __restrict__ b1,
                          const float* __restrict__ w2, const float* __restrict__ b2,
                          const float* __restrict__ mask, float* __restrict__ kerns) {
  int b = blockIdx.x / 9, r = blockIdx.x % 9;
  int t = threadIdx.x;                          // sim channel 0..63
  float p4 = 0.f, p5 = 0.f;
  for (int c = 0; c < 256; ++c) p4 += pool4[(size_t)(b * 256 + c) * 9 + r] * ws4[t * 256 + c];
  for (int c = 0; c < 512; ++c) p5 += pool5[(size_t)(b * 512 + c) * 9 + r] * ws5[t * 512 + c];
  __shared__ float red[64];
  red[t] = p4 * p5; __syncthreads();
  for (int st = 32; st > 0; st >>= 1) {
    if (t < st) red[t] += red[t + st];
    __syncthreads();
  }
  if (t == 0) {
    float g = red[0] * (1.f / (1.f + __expf(-mask[r])));
    float h[16], lg[9], mx = -1e30f;
    for (int j = 0; j < 16; ++j) { float v = g * w1[j] + b1[j]; h[j] = v > 0.f ? v : 0.f; }
    for (int l = 0; l < 9; ++l) {
      float s = b2[l];
      for (int j = 0; j < 16; ++j) s += h[j] * w2[l * 16 + j];
      lg[l] = s; if (s > mx) mx = s;
    }
    float den = 0.f;
    for (int l = 0; l < 9; ++l) { lg[l] = __expf(lg[l] - mx); den += lg[l]; }
    for (int l = 0; l < 9; ++l) kerns[b * 81 + r * 9 + l] = lg[l] / den;
  }
}

// ---------------------------------------------------------------------------
// GEMM-1: implicit-im2col  fused[M=32768, N=256] = A[M, 2816] * Wcat[2816, 256]
// Block 128x128, 8 waves (4M x 2N), wave tile 32x64 = 2x4 WMMA 16x16x32 f16.
// Register double-buffered staging: global loads of tile k+1 issue before the
// WMMAs of tile k; LDS ping-pong removes one barrier per K-step.
// ---------------------------------------------------------------------------
__global__ __launch_bounds__(256) void k_gemm1(const float* __restrict__ c4,
                                               const float* __restrict__ c5,
                                               const h16* __restrict__ wcat,
                                               float* __restrict__ fused) {
  __shared__ alignas(32) h16 Ash[2][8 * 512];
  const int tid = threadIdx.x;
  const int lane = tid & 31, wid = tid >> 5;
  const int wm = wid & 3, wn = wid >> 2;
  const int m0 = blockIdx.x * 128;
  const int n0 = blockIdx.y * 128;
  const int b = m0 >> 12;
  const int pixBase = m0 & 4095;
  const int kkc = tid & 31;                 // K-within-chunk: constant per thread
  const int rowBase = tid >> 5;             // 0..7
  // A-fragment store position (constant per thread except the M part)
  int laOff, eIdx;
  if (kkc < 16)      { laOff = 0;  eIdx = kkc;      }
  else               { laOff = 16; eIdx = kkc - 16; }

  float vreg[16];
  auto load_tile = [&](int kb) {
    int k = (kb << 5) + kkc;
    if (k < 2304) {                                    // c4 3x3 im2col
      int cc = k / 9, tt = k - cc * 9;
      int dy = tt / 3 - 1, dx = tt % 3 - 1;
      const float* base = c4 + (((size_t)(b * 256 + cc)) << 12);
      #pragma unroll
      for (int i = 0; i < 16; ++i) {
        int pix = pixBase + i * 8 + rowBase;
        int y = (pix >> 6) + dy, x = (pix & 63) + dx;
        vreg[i] = (y >= 0 && y < 64 && x >= 0 && x < 64) ? base[(y << 6) + x] : 0.f;
      }
    } else {                                           // c5 nearest-upsample, 1x1
      int cc = k - 2304;
      const float* base = c5 + (((size_t)(b * 512 + cc)) << 10);
      #pragma unroll
      for (int i = 0; i < 16; ++i) {
        int pix = pixBase + i * 8 + rowBase;
        vreg[i] = base[(((pix >> 6) >> 1) << 5) + ((pix & 63) >> 1)];
      }
    }
  };
  auto store_tile = [&](h16* buf) {
    #pragma unroll
    for (int i = 0; i < 16; ++i) {
      int mLoc = i * 8 + rowBase;
      buf[(mLoc >> 4) * 512 + ((mLoc & 15) + laOff) * 16 + eIdx] = (h16)vreg[i];
    }
  };

  v8f acc[2][4] = {};
  load_tile(0);
  store_tile(Ash[0]);
  __syncthreads();
  for (int kb = 0; kb < 88; ++kb) {
    const h16* cur = Ash[kb & 1];
    if (kb + 1 < 88) load_tile(kb + 1);      // in-flight during WMMAs
    v16h a0 = *(const v16h*)&cur[(wm * 2 + 0) * 512 + lane * 16];
    v16h a1 = *(const v16h*)&cur[(wm * 2 + 1) * 512 + lane * 16];
    #pragma unroll
    for (int j = 0; j < 4; ++j) {
      int nbG = blockIdx.y * 8 + wn * 4 + j;
      v16h bf = *(const v16h*)&wcat[((size_t)(kb * 16 + nbG)) * 512 + lane * 16];
      if (kb + 1 < 88)
        __builtin_prefetch(&wcat[((size_t)((kb + 1) * 16 + nbG)) * 512 + lane * 16], 0, 3);
      acc[0][j] = __builtin_amdgcn_wmma_f32_16x16x32_f16(false, a0, false, bf,
                                                         (short)0, acc[0][j], false, false);
      acc[1][j] = __builtin_amdgcn_wmma_f32_16x16x32_f16(false, a1, false, bf,
                                                         (short)0, acc[1][j], false, false);
    }
    if (kb + 1 < 88) store_tile(Ash[(kb + 1) & 1]);
    __syncthreads();
  }
  const int laneN = lane & 15, hiM = (lane >> 4) * 8;
  #pragma unroll
  for (int i = 0; i < 2; ++i)
    #pragma unroll
    for (int j = 0; j < 4; ++j)
      #pragma unroll
      for (int r = 0; r < 8; ++r) {
        int Mg = m0 + wm * 32 + i * 16 + r + hiM;
        int Ng = n0 + wn * 64 + j * 16 + laneN;
        fused[(size_t)Mg * 256 + Ng] = acc[i][j][r];
      }
}

// ---------------------------------------------------------------------------
// GEMM-2: ybig[M=32768, N=512] = fused[M,256] * [w_to_fuse^T | Wcomb^T]
// A-tiles staged raw (f32) into LDS via async global->LDS (double-buffered,
// ASYNCcnt-tracked); A fragments built from LDS with 16B ds loads + pk cvt.
// Row pitch 36 floats keeps the 16 fragment lanes on distinct banks.
// ---------------------------------------------------------------------------
__global__ __launch_bounds__(256) void k_gemm2(const float* __restrict__ fused,
                                               const h16* __restrict__ wbig,
                                               float* __restrict__ ybig) {
  __shared__ alignas(16) float raw[2][128 * 36];
  const int tid = threadIdx.x;
  const int lane = tid & 31, wid = tid >> 5;
  const int wm = wid & 3, wn = wid >> 2;
  const int m0 = blockIdx.x * 128;
  const int n0 = blockIdx.y * 128;
  const int rowB = tid >> 3;                // 0..31
  const int chunk = (tid & 7) * 4;          // col base 0,4,...,28

  auto issue = [&](int kb) {
    float* dst = raw[kb & 1];
    #pragma unroll
    for (int p = 0; p < 4; ++p) {
      int row = rowB + p * 32;
      const float* g = fused + (size_t)(m0 + row) * 256 + (kb << 5) + chunk;
      float* l = dst + row * 36 + chunk;
#if GL_ASYNC
      __builtin_amdgcn_global_load_async_to_lds_b128(
          (gasp)(uintptr_t)g, (lasp)(uintptr_t)(uint32_t)(uintptr_t)l, 0, 0);
#else
      *(v4f*)l = *(const v4f*)g;
#endif
    }
  };

  v8f acc[2][4] = {};
  issue(0);
  for (int kb = 0; kb < 8; ++kb) {
    WAIT_ASYNC();                           // tile kb landed in LDS
    __syncthreads();
    if (kb + 1 < 8) issue(kb + 1);          // overlaps with WMMAs below
    const float* cur = raw[kb & 1];
    const int Ml = lane & 15, hi = lane >> 4;
    const int r0 = hi ? 16 : 0, r1 = hi ? 24 : 8;
    v16h afr[2];
    #pragma unroll
    for (int i = 0; i < 2; ++i) {
      const float* rp = cur + ((wm * 2 + i) * 16 + Ml) * 36;
      v4f f0 = *(const v4f*)(rp + r0);
      v4f f1 = *(const v4f*)(rp + r0 + 4);
      v4f f2 = *(const v4f*)(rp + r1);
      v4f f3 = *(const v4f*)(rp + r1 + 4);
      v16h a;
      #pragma unroll
      for (int j = 0; j < 4; ++j) {
        a[j]      = (h16)f0[j];
        a[4 + j]  = (h16)f1[j];
        a[8 + j]  = (h16)f2[j];
        a[12 + j] = (h16)f3[j];
      }
      afr[i] = a;
    }
    #pragma unroll
    for (int j = 0; j < 4; ++j) {
      int nbG = blockIdx.y * 8 + wn * 4 + j;
      v16h bf = *(const v16h*)&wbig[((size_t)(kb * 32 + nbG)) * 512 + lane * 16];
      if (kb + 1 < 8)
        __builtin_prefetch(&wbig[((size_t)((kb + 1) * 32 + nbG)) * 512 + lane * 16], 0, 3);
      acc[0][j] = __builtin_amdgcn_wmma_f32_16x16x32_f16(false, afr[0], false, bf,
                                                         (short)0, acc[0][j], false, false);
      acc[1][j] = __builtin_amdgcn_wmma_f32_16x16x32_f16(false, afr[1], false, bf,
                                                         (short)0, acc[1][j], false, false);
    }
  }
  const int laneN = lane & 15, hiM = (lane >> 4) * 8;
  #pragma unroll
  for (int i = 0; i < 2; ++i)
    #pragma unroll
    for (int j = 0; j < 4; ++j)
      #pragma unroll
      for (int r = 0; r < 8; ++r) {
        int Mg = m0 + wm * 32 + i * 16 + r + hiM;
        int Ng = n0 + wn * 64 + j * 16 + laneN;
        ybig[(size_t)Mg * 512 + Ng] = acc[i][j][r];
      }
}

// Deterministic two-stage BN statistics over y (= ybig cols 0..255)
__global__ void k_bnstats(const float* __restrict__ ybig, float* __restrict__ part) {
  int ch = threadIdx.x;
  int m0 = blockIdx.x * 128;
  float s = 0.f, s2 = 0.f;
  for (int r = 0; r < 128; ++r) {
    float v = ybig[(size_t)(m0 + r) * 512 + ch];
    s += v; s2 += v * v;
  }
  part[blockIdx.x * 256 + ch] = s;
  part[65536 + blockIdx.x * 256 + ch] = s2;
}

__global__ void k_bnfinal(const float* __restrict__ part, float* __restrict__ stat) {
  int c = threadIdx.x;
  float s = 0.f, s2 = 0.f;
  for (int mb = 0; mb < 256; ++mb) {
    s  += part[mb * 256 + c];
    s2 += part[65536 + mb * 256 + c];
  }
  float mu  = s * (1.f / 32768.f);
  float var = s2 * (1.f / 32768.f) - mu * mu;
  stat[c] = mu;
  stat[256 + c] = rsqrtf(var + 1e-5f);
}

// Fused epilogue: BN -> SiLU -> spatially-varying 3x3 filter -> + fused_red
__global__ __launch_bounds__(256) void k_out(const float* __restrict__ ybig,
                                             const float* __restrict__ stat,
                                             const float* __restrict__ gamma,
                                             const float* __restrict__ beta,
                                             const float* __restrict__ kerns,
                                             float* __restrict__ out) {
  int pix = blockIdx.x;                        // b*4096 + h*64 + w
  int b = pix >> 12, hw = pix & 4095, h = hw >> 6, w = hw & 63;
  int reg = ((h * 3) >> 6) * 3 + ((w * 3) >> 6);
  __shared__ float kc[9];
  if (threadIdx.x < 9) kc[threadIdx.x] = kerns[b * 81 + reg * 9 + threadIdx.x];
  __syncthreads();
  int c = threadIdx.x;
  float mu = stat[c], inv = stat[256 + c], ga = gamma[c], be = beta[c];
  float acc = 0.f;
  #pragma unroll
  for (int t = 0; t < 9; ++t) {
    int hh = h + t / 3 - 1, ww = w + t % 3 - 1;
    if (hh >= 0 && hh < 64 && ww >= 0 && ww < 64) {
      float yv = ybig[(size_t)((b << 12) + (hh << 6) + ww) * 512 + c];
      float v = (yv - mu) * inv * ga + be;
      acc += kc[t] * (v / (1.f + __expf(-v)));   // SiLU
    }
  }
  acc += ybig[(size_t)pix * 512 + 256 + c];      // fused_red residual
  out[(((size_t)(b * 256 + c)) << 12) + hw] = acc;
}

extern "C" void kernel_launch(void* const* d_in, const int* in_sizes, int n_in,
                              void* d_out, int out_size, void* d_ws, size_t ws_size,
                              hipStream_t stream) {
  (void)in_sizes; (void)n_in; (void)out_size; (void)ws_size;
  const float* c4        = (const float*)d_in[0];
  const float* c5        = (const float*)d_in[1];
  const float* w_to_fuse = (const float*)d_in[2];
  const float* bn_gamma  = (const float*)d_in[3];
  const float* bn_beta   = (const float*)d_in[4];
  const float* w_c4_proc = (const float*)d_in[5];
  const float* w_conv1   = (const float*)d_in[6];
  const float* w_reshape = (const float*)d_in[7];
  const float* w_proj    = (const float*)d_in[8];
  const float* w_sim4    = (const float*)d_in[9];
  const float* w_sim5    = (const float*)d_in[10];
  const float* kg_w1     = (const float*)d_in[11];
  const float* kg_b1     = (const float*)d_in[12];
  const float* kg_w2     = (const float*)d_in[13];
  const float* kg_b2     = (const float*)d_in[14];
  const float* mask_raw  = (const float*)d_in[15];

  char* ws = (char*)d_ws;
  h16*   wcat   = (h16*)(ws + OFF_WCAT);
  h16*   wbig   = (h16*)(ws + OFF_WBIG);
  float* pool4  = (float*)(ws + OFF_POOL4);
  float* pool5  = (float*)(ws + OFF_POOL5);
  float* kerns  = (float*)(ws + OFF_KERNS);
  float* bnpart = (float*)(ws + OFF_BNPART);
  float* bnstat = (float*)(ws + OFF_BNSTAT);
  float* fused  = (float*)(ws + OFF_FUSED);
  float* ybig   = (float*)(ws + OFF_YBIG);
  float* outp   = (float*)d_out;

  k_pack_wcat<<<2816, 256, 0, stream>>>(w_c4_proc, w_conv1, wcat);
  k_pack_wbig<<<512, 256, 0, stream>>>(w_to_fuse, w_reshape, w_proj, wbig);
  k_pool4<<<8 * 256 * 9, 64, 0, stream>>>(c4, pool4);
  k_pool5<<<8 * 512 * 9, 64, 0, stream>>>(c5, pool5);
  k_sim_mlp<<<72, 64, 0, stream>>>(pool4, pool5, w_sim4, w_sim5,
                                   kg_w1, kg_b1, kg_w2, kg_b2, mask_raw, kerns);
  k_gemm1<<<dim3(256, 2), 256, 0, stream>>>(c4, c5, wcat, fused);
  k_gemm2<<<dim3(256, 4), 256, 0, stream>>>(fused, wbig, ybig);
  k_bnstats<<<256, 256, 0, stream>>>(ybig, bnpart);
  k_bnfinal<<<1, 256, 0, stream>>>(bnpart, bnstat);
  k_out<<<32768, 256, 0, stream>>>(ybig, bnstat, bn_gamma, bn_beta, kerns, outp);
}
